// Attention_11192684774063
// MI455X (gfx1250) — matrix operand order repaired and checked by reading, not compiled, not run on hardware
//
#include <hip/hip_runtime.h>

// Attention: Q,K,V [B=8, C=256, N=2048] fp32.
// scores = Q^T K / sqrt(N); p = softmax(scores); out = einsum(bnm,bcm->bcn).
// d_out = out (B*C*N floats) ++ p_attn (B*N*N floats).

#define BB 8
#define CC 256
#define NN 2048

typedef __bf16 v16bf __attribute__((ext_vector_type(16)));
typedef __bf16 v2bf  __attribute__((ext_vector_type(2)));
typedef float  v8f   __attribute__((ext_vector_type(8)));
typedef float  v4f   __attribute__((ext_vector_type(4)));
typedef unsigned int v4u __attribute__((ext_vector_type(4)));
typedef unsigned int v8u __attribute__((ext_vector_type(8)));

// pack two f32 -> one dword {bf16(lo), bf16(hi)} (lowers to v_cvt_pk_bf16_f32)
__device__ __forceinline__ unsigned int pk_bf16(float lo, float hi) {
  v2bf t; t[0] = (__bf16)lo; t[1] = (__bf16)hi;
  return __builtin_bit_cast(unsigned int, t);
}

// two b128 LDS words -> one 16x32 bf16 fragment (VGPR v = word v)
__device__ __forceinline__ v16bf frag_from_words(v4u lo, v4u hi) {
  v8u w;
  w[0] = lo[0]; w[1] = lo[1]; w[2] = lo[2]; w[3] = lo[3];
  w[4] = hi[0]; w[5] = hi[1]; w[6] = hi[2]; w[7] = hi[3];
  return __builtin_bit_cast(v16bf, w);
}

__global__ __launch_bounds__(256)
void attn_fused_kernel(const float* __restrict__ Q,
                       const float* __restrict__ K,
                       const float* __restrict__ V,
                       float* __restrict__ Out,
                       float* __restrict__ P) {
  // 64 KB LDS (dwords):
  //  sQw  = words [0..2112)        packed-bf16 Q block, pitch 132/row (RESIDENT in phase 1)
  //  slice(w) = 2112 + w*1088      per-wave packed-bf16 K tile, pitch 68/row
  //  red  = floats at word 10816   (8 waves x 16 rows partials)
  //  sP (phase 2) = words [0..2112) packed-bf16 P chunk, pitch 132/row (sQw dead by then)
  __shared__ unsigned int smem[16384];

  const int b    = blockIdx.y;
  const int n0   = blockIdx.x * 16;       // 16 query rows per workgroup
  const int tid  = threadIdx.x;
  const int wave = tid >> 5;
  const int lane = tid & 31;
  const int half = lane >> 4;
  const int lcol = lane & 15;
  const int laneJ = lane >> 2;            // staging: word-column within octet
  const int laneM = lane & 3;             // staging: m-quad
  const float scale = 0.022097086912079608f; // 1/sqrt(2048)  (d_k = N, faithful)

  const size_t bOff = (size_t)b * CC * NN;
  unsigned int* sQw = smem;
  unsigned int* sKw = smem + 2112 + wave * 1088;
  float*        red = (float*)(smem + 10816);

  // ---- stage Q block as packed bf16 [r][c/2] (scale folded), pitch 132 ----
  #pragma unroll
  for (int i = 0; i < 2; ++i) {
    int p = tid + 256 * i;               // 512 pair-slots
    int j = p >> 2, rq = p & 3;          // j = c/2 (0..127)
    const float* gp = Q + bOff + (size_t)(2 * j) * NN + n0 + rq * 4;
    v4f a = *(const v4f*)gp;
    v4f c = *(const v4f*)(gp + NN);
    #pragma unroll
    for (int q = 0; q < 4; ++q)
      sQw[(rq * 4 + q) * 132 + j] = pk_bf16(a[q] * scale, c[q] * scale);
  }
  __syncthreads();

  // ---- phase 1: scores; wave w owns cols [256w, 256w+256); h-outer, pipelined
  v8f acc[16];
  #pragma unroll
  for (int t = 0; t < 16; ++t) {
    v8f z = {0.f, 0.f, 0.f, 0.f, 0.f, 0.f, 0.f, 0.f};
    acc[t] = z;
  }

  const float* kbase = K + bOff + (size_t)(2 * laneJ) * NN + wave * 256 + laneM * 4;
  v4f cur[16];
  // prologue: load tile (h=0, t=0)
  #pragma unroll
  for (int i = 0; i < 8; ++i) {
    const float* gp = kbase + (size_t)(16 * i) * NN;
    cur[2 * i]     = *(const v4f*)gp;
    cur[2 * i + 1] = *(const v4f*)(gp + NN);
  }

  #pragma unroll
  for (int h = 0; h < 2; ++h) {
    // A fragments for this c-half from resident sQw: 2x ds_load_b128 each
    v16bf aQh[4];
    #pragma unroll
    for (int kfl = 0; kfl < 4; ++kfl) {
      const unsigned int* ap = sQw + lcol * 132 + (h * 4 + kfl) * 16 + half * 8;
      aQh[kfl] = frag_from_words(*(const v4u*)ap, *(const v4u*)(ap + 4));
    }
    #pragma unroll
    for (int t = 0; t < 16; ++t) {
      // consume cur -> packed-bf16 slice [m][c/2], pitch 68 (conflict-free stores)
      #pragma unroll
      for (int i = 0; i < 8; ++i) {
        #pragma unroll
        for (int q = 0; q < 4; ++q)
          sKw[(laneM * 4 + q) * 68 + laneJ + 8 * i] =
              pk_bf16(cur[2 * i][q], cur[2 * i + 1][q]);
      }
      // prefetch next step's tile while this step's LDS reads + WMMAs run
      if (!(h == 1 && t == 15)) {
        const int nh = (t == 15) ? 1 : h;
        const int nt = (t == 15) ? 0 : t + 1;
        #pragma unroll
        for (int i = 0; i < 8; ++i) {
          const float* gp = kbase + (size_t)(nh * 128 + 16 * i) * NN + nt * 16;
          cur[2 * i]     = *(const v4f*)gp;
          cur[2 * i + 1] = *(const v4f*)(gp + NN);
        }
      }
      // B fragments (2x ds_load_b128 each), pipelined one kfl ahead of WMMA
      const unsigned int* fpb = sKw + lcol * 68 + half * 8;
      v16bf bK = frag_from_words(*(const v4u*)fpb, *(const v4u*)(fpb + 4));
      #pragma unroll
      for (int kfl = 0; kfl < 4; ++kfl) {
        v16bf bKn;
        if (kfl < 3) {
          const unsigned int* fp = fpb + (kfl + 1) * 16;
          bKn = frag_from_words(*(const v4u*)fp, *(const v4u*)(fp + 4));
        }
        acc[t] = __builtin_amdgcn_wmma_f32_16x16x32_bf16(
            false, aQh[kfl], false, bK, (short)0, acc[t], false, false);
        bK = bKn;
      }
    }
  }
  __syncthreads();  // all phase-1 LDS use done

  // ---- softmax (exact, f32). C layout: row = v+8*half, col = m0+lcol ----
  float gmax[8], inv[8];
  {
    float mx[8];
    #pragma unroll
    for (int v = 0; v < 8; ++v) {
      float m = acc[0][v];
      #pragma unroll
      for (int t = 1; t < 16; ++t) m = fmaxf(m, acc[t][v]);
      mx[v] = m;
    }
    #pragma unroll
    for (int off = 8; off; off >>= 1) {
      #pragma unroll
      for (int v = 0; v < 8; ++v)
        mx[v] = fmaxf(mx[v], __shfl_xor(mx[v], off, 32));
    }
    if (lcol == 0) {
      #pragma unroll
      for (int v = 0; v < 8; ++v) red[wave * 16 + v + 8 * half] = mx[v];
    }
  }
  __syncthreads();
  #pragma unroll
  for (int v = 0; v < 8; ++v) {
    float m = red[v + 8 * half];
    #pragma unroll
    for (int w = 1; w < 8; ++w) m = fmaxf(m, red[w * 16 + v + 8 * half]);
    gmax[v] = m;
  }
  __syncthreads();
  {
    float sm[8];
    #pragma unroll
    for (int v = 0; v < 8; ++v) sm[v] = 0.f;
    #pragma unroll
    for (int t = 0; t < 16; ++t) {
      #pragma unroll
      for (int v = 0; v < 8; ++v) {
        float e = __expf(acc[t][v] - gmax[v]);
        acc[t][v] = e;
        sm[v] += e;
      }
    }
    #pragma unroll
    for (int off = 8; off; off >>= 1) {
      #pragma unroll
      for (int v = 0; v < 8; ++v)
        sm[v] += __shfl_xor(sm[v], off, 32);
    }
    if (lcol == 0) {
      #pragma unroll
      for (int v = 0; v < 8; ++v) red[wave * 16 + v + 8 * half] = sm[v];
    }
  }
  __syncthreads();
  #pragma unroll
  for (int v = 0; v < 8; ++v) {
    float s = 0.f;
    #pragma unroll
    for (int w = 0; w < 8; ++w) s += red[w * 16 + v + 8 * half];
    inv[v] = 1.0f / s;
  }

  // normalize; stream p_attn non-temporally (never re-read; keep L2 for Q/K/V)
  {
    float* pAt = P + (size_t)b * NN * NN + (size_t)n0 * NN;
    #pragma unroll
    for (int t = 0; t < 16; ++t) {
      const int col = wave * 256 + t * 16 + lcol;
      #pragma unroll
      for (int v = 0; v < 8; ++v) {
        float p = acc[t][v] * inv[v];
        acc[t][v] = p;
        __builtin_nontemporal_store(p, pAt + (size_t)(v + 8 * half) * NN + col);
      }
    }
  }

  // ---- phase 2: O[16 x 256] = P(16x2048) * V^T, chunked over m, pipelined
  __bf16* sP = (__bf16*)smem;            // [row][264 bf16] == [row][132 words]
  v8f oacc[2];
  {
    v8f z = {0.f, 0.f, 0.f, 0.f, 0.f, 0.f, 0.f, 0.f};
    oacc[0] = z; oacc[1] = z;
  }
  const int c0 = wave * 32;              // 2 c-tiles per wave
  const float* vb0 = V + bOff + (size_t)(c0 + lcol) * NN + half * 16;   // s=0
  const float* vb1 = vb0 + (size_t)16 * NN;                             // s=1
  const unsigned int* apb = smem + lcol * 132 + half * 8;

  for (int j = 0; j < 8; ++j) {          // chunk m in [256j, 256j+256), owner = wave j
    __syncthreads();
    if (wave == j) {
      #pragma unroll
      for (int t = 0; t < 16; ++t) {
        #pragma unroll
        for (int v = 0; v < 8; ++v)
          sP[(v + 8 * half) * 264 + t * 16 + lcol] = (__bf16)acc[t][v];
      }
    }
    __syncthreads();
    // preload V for kf=0 (both c-tiles): 8x global_load_b128, one wait
    v4f vb[8];
    #pragma unroll
    for (int q = 0; q < 4; ++q) {
      vb[q]     = *(const v4f*)(vb0 + j * 256 + q * 4);
      vb[4 + q] = *(const v4f*)(vb1 + j * 256 + q * 4);
    }
    // preload A fragment for kf=0
    v16bf aP = frag_from_words(*(const v4u*)apb, *(const v4u*)(apb + 4));
    #pragma unroll
    for (int kf = 0; kf < 8; ++kf) {
      // pipeline next A fragment ahead of this kf's WMMAs
      v16bf aPn;
      if (kf < 7) {
        const unsigned int* ap = apb + (kf + 1) * 16;
        aPn = frag_from_words(*(const v4u*)ap, *(const v4u*)(ap + 4));
      }
      // convert current V regs to bf16 fragments
      v16bf bV0, bV1;
      #pragma unroll
      for (int q = 0; q < 4; ++q) {
        #pragma unroll
        for (int e = 0; e < 4; ++e) {
          bV0[4 * q + e] = (__bf16)vb[q][e];
          bV1[4 * q + e] = (__bf16)vb[4 + q][e];
        }
      }
      // prefetch next kf's V while the two WMMAs run
      if (kf < 7) {
        #pragma unroll
        for (int q = 0; q < 4; ++q) {
          vb[q]     = *(const v4f*)(vb0 + j * 256 + (kf + 1) * 32 + q * 4);
          vb[4 + q] = *(const v4f*)(vb1 + j * 256 + (kf + 1) * 32 + q * 4);
        }
      }
      oacc[0] = __builtin_amdgcn_wmma_f32_16x16x32_bf16(
          false, aP, false, bV0, (short)0, oacc[0], false, false);
      oacc[1] = __builtin_amdgcn_wmma_f32_16x16x32_bf16(
          false, aP, false, bV1, (short)0, oacc[1], false, false);
      aP = aPn;
    }
  }

  // ---- write O: out[b, c, n0+row] (streamed, non-temporal)
  #pragma unroll
  for (int s = 0; s < 2; ++s) {
    #pragma unroll
    for (int v = 0; v < 8; ++v)
      __builtin_nontemporal_store(
          oacc[s][v],
          Out + bOff + (size_t)(c0 + s * 16 + lcol) * NN + n0 + v + 8 * half);
  }
}

extern "C" void kernel_launch(void* const* d_in, const int* in_sizes, int n_in,
                              void* d_out, int out_size, void* d_ws, size_t ws_size,
                              hipStream_t stream) {
  (void)in_sizes; (void)n_in; (void)d_ws; (void)ws_size; (void)out_size;
  const float* Q = (const float*)d_in[0];
  const float* K = (const float*)d_in[1];
  const float* V = (const float*)d_in[2];
  // d_in[3] = mask scalar (0) -> dropout branch skipped; ignored.
  float* out = (float*)d_out;
  float* p   = out + (size_t)BB * CC * NN;   // p_attn follows out
  dim3 grid(NN / 16, BB), block(256);
  attn_fused_kernel<<<grid, block, 0, stream>>>(Q, K, V, out, p);
}
